// Attention_21990232555717
// MI455X (gfx1250) — compile-verified, matrix operand order checked by reading, not run on hardware
//
#include <hip/hip_runtime.h>
#include <hip/hip_bf16.h>

typedef __bf16 bf16_t;
typedef __attribute__((ext_vector_type(8)))  __bf16 v8bf;
typedef __attribute__((ext_vector_type(16))) __bf16 v16bf;
typedef __attribute__((ext_vector_type(8)))  float  v8f;
typedef __attribute__((ext_vector_type(4)))  unsigned int v4u;
typedef __attribute__((ext_vector_type(8)))  int v8i_;
typedef __attribute__((ext_vector_type(4)))  int v4i_;

#define WMMA_BF16(a, b, c) \
  __builtin_amdgcn_wmma_f32_16x16x32_bf16(false, (a), false, (b), (short)0, (c), false, false)

// ---------------------------------------------------------------------------
// TDM: issue a 2D bf16 tile load (tile_d0 elems/row x tile_d1 rows) from a
// row-major tensor (row stride = stride_elems) into contiguous LDS at
// lds_byte_off. One instruction per wave; tracked by TENSORcnt.
// D# layout per CDNA5 ISA ch.8 (group0: count/lds/global/type, group1:
// data_size + dims + tile dims + dim0 stride). Groups 2/3 zero => 2D tensor.
// This toolchain exposes the 6-arg builtin: (g0 v4u, g1 v8i, v4i, v4i, v8i, cpol).
// ---------------------------------------------------------------------------
__device__ __forceinline__ void tdm_load_2d(unsigned lds_byte_off, const void* gptr,
                                            unsigned tile_d0, unsigned tile_d1,
                                            unsigned tensor_d0, unsigned tensor_d1,
                                            unsigned long long stride_elems) {
  unsigned long long ga = (unsigned long long)gptr;
  v4u g0;
  g0[0] = 1u;                                            // count=1, user D#
  g0[1] = lds_byte_off;                                  // lds_addr (bytes)
  g0[2] = (unsigned)(ga & 0xffffffffu);                  // global_addr[31:0]
  g0[3] = (unsigned)((ga >> 32) & 0x1ffffffu) | (2u << 30);  // addr[56:32] | type=2
  v8i_ g1;
  g1[0] = (int)(1u << 16);                               // data_size=1 (2 bytes)
  g1[1] = (int)((tensor_d0 & 0xffffu) << 16);            // tensor_dim0[15:0]
  g1[2] = (int)((tensor_d0 >> 16) | ((tensor_d1 & 0xffffu) << 16));
  g1[3] = (int)((tensor_d1 >> 16) | (tile_d0 << 16));    // tile_dim0
  g1[4] = (int)tile_d1;                                  // tile_dim1 (tile_dim2=0)
  g1[5] = (int)(unsigned)(stride_elems & 0xffffffffull); // dim0_stride low
  g1[6] = (int)(unsigned)((stride_elems >> 32) & 0xffffull);
  g1[7] = 0;
  v4i_ z4 = {0, 0, 0, 0};
  v8i_ z8 = {0, 0, 0, 0, 0, 0, 0, 0};
  __builtin_amdgcn_tensor_load_to_lds(g0, g1, z4, z4, z8, 0);
}

// Load a 16-lane-row x 32-deep bf16 WMMA fragment.
// Lane l (l<16) holds row (l&15), K slots {kb..kb+7, kb+16..kb+23}, kb=(l>>4)*8.
// Works for A (rows = M) and for B when the K x N matrix is stored N-major.
__device__ __forceinline__ v16bf load_frag(const bf16_t* base, int stride_halves) {
  const int lane = threadIdx.x & 31;
  const int row  = lane & 15;
  const int kb   = (lane >> 4) << 3;
  const bf16_t* p = base + (size_t)row * stride_halves + kb;
  v8bf lo = *(const v8bf*)(p);
  v8bf hi = *(const v8bf*)(p + 16);
  v16bf r;
#pragma unroll
  for (int i = 0; i < 8; ++i) { r[i] = lo[i]; r[i + 8] = hi[i]; }
  return r;
}

// ---------------- conversion kernels ----------------
__global__ void cvt_bf16(const float* __restrict__ s, bf16_t* __restrict__ d, int n) {
  int i = blockIdx.x * blockDim.x + threadIdx.x;
  if (i < n) d[i] = (bf16_t)s[i];
}

// s is [R][C] fp32 row-major; d becomes [C][R] bf16 (transposed)
__global__ void cvt_bf16_T(const float* __restrict__ s, bf16_t* __restrict__ d, int R, int C) {
  int i = blockIdx.x * blockDim.x + threadIdx.x;
  if (i < R * C) {
    int r = i / C, c = i % C;
    d[(size_t)c * R + r] = (bf16_t)s[i];
  }
}

// ---------------- bf16 WMMA GEMM with TDM-fed LDS tiles ----------------
// C[M][N] = A[M][K] * B[K][N], with B supplied transposed as BT[N][K].
// MODE 0: scatter epilogue into Q [a][h][n][64], K [a][h][n][64], V^T [a][h][64][n]
// MODE 1: fp32 out + bias
#define BM 64
#define BN 64
#define BK 32

template <int MODE>
__global__ __launch_bounds__(128) void gemm_bf16(
    const bf16_t* __restrict__ A, const bf16_t* __restrict__ BT,
    int M, int N, int K,
    bf16_t* __restrict__ qout, bf16_t* __restrict__ kout, bf16_t* __restrict__ vtout,
    float* __restrict__ fout, const float* __restrict__ bias) {
  __shared__ __align__(16) bf16_t sA[2][BM][BK];
  __shared__ __align__(16) bf16_t sB[2][BN][BK];

  const int m0 = blockIdx.y * BM;
  const int n0 = blockIdx.x * BN;
  const int tid = threadIdx.x;
  const int wave = tid >> 5, lane = tid & 31;
  const int wm = (wave >> 1) * 32, wn = (wave & 1) * 32;
  const int nk = K / BK;

  // prologue: kick off tile 0 via the Tensor Data Mover
  if (wave == 0) {
    tdm_load_2d((unsigned)(size_t)&sA[0][0][0], A + (size_t)m0 * K,
                BK, BM, (unsigned)K, (unsigned)M, (unsigned long long)K);
    tdm_load_2d((unsigned)(size_t)&sB[0][0][0], BT + (size_t)n0 * K,
                BK, BN, (unsigned)K, (unsigned)N, (unsigned long long)K);
  }

  v8f acc[2][2] = {};

  for (int i = 0; i < nk; ++i) {
    if (wave == 0) {
      if (i + 1 < nk) {  // double-buffer: issue next pair, wait for current pair
        const int k1 = (i + 1) * BK;
        tdm_load_2d((unsigned)(size_t)&sA[(i + 1) & 1][0][0], A + (size_t)m0 * K + k1,
                    BK, BM, (unsigned)K, (unsigned)M, (unsigned long long)K);
        tdm_load_2d((unsigned)(size_t)&sB[(i + 1) & 1][0][0], BT + (size_t)n0 * K + k1,
                    BK, BN, (unsigned)K, (unsigned)N, (unsigned long long)K);
        __builtin_amdgcn_s_wait_tensorcnt(2);  // TDM in-order: older pair done
      } else {
        __builtin_amdgcn_s_wait_tensorcnt(0);
      }
    }
    __syncthreads();
    const int b = i & 1;
    v16bf a0 = load_frag(&sA[b][wm][0], BK);
    v16bf a1 = load_frag(&sA[b][wm + 16][0], BK);
    v16bf b0 = load_frag(&sB[b][wn][0], BK);
    v16bf b1 = load_frag(&sB[b][wn + 16][0], BK);
    acc[0][0] = WMMA_BF16(a0, b0, acc[0][0]);
    acc[0][1] = WMMA_BF16(a0, b1, acc[0][1]);
    acc[1][0] = WMMA_BF16(a1, b0, acc[1][0]);
    acc[1][1] = WMMA_BF16(a1, b1, acc[1][1]);
    __syncthreads();  // compute done before TDM overwrites this buffer
  }

  // C/D layout: VGPR r -> M = r (lanes 0-15) or r+8 (lanes 16-31); N = lane&15
  const int lrow = (lane >> 4) * 8;
  const int lcol = lane & 15;
#pragma unroll
  for (int i = 0; i < 2; ++i)
#pragma unroll
    for (int j = 0; j < 2; ++j)
#pragma unroll
      for (int r = 0; r < 8; ++r) {
        int gm = m0 + wm + i * 16 + r + lrow;
        int gn = n0 + wn + j * 16 + lcol;
        float v = acc[i][j][r];
        if (MODE == 0) {
          int aa = gm >> 10, tok = gm & 1023;
          if (gn < 1024) {
            int h = gn >> 6, dd = gn & 63;
            qout[(((size_t)(aa * 16 + h)) * 1024 + tok) * 64 + dd] = (bf16_t)v;
          } else if (gn < 2048) {
            int c = gn - 1024, h = c >> 6, dd = c & 63;
            kout[(((size_t)(aa * 16 + h)) * 1024 + tok) * 64 + dd] = (bf16_t)v;
          } else {
            int c = gn - 2048, h = c >> 6, dd = c & 63;
            vtout[(((size_t)(aa * 16 + h)) * 64 + dd) * 1024 + tok] = (bf16_t)v;
          }
        } else {
          fout[(size_t)gm * N + gn] = v + bias[gn];
        }
      }
}

// ---------------- fused attention (softmax over HEADS axis) ----------------
#define NH 16
#define HD 64
#define SEQ 1024

__global__ __launch_bounds__(256) void attn_fused(
    const bf16_t* __restrict__ Q, const bf16_t* __restrict__ Kb,
    const bf16_t* __restrict__ VT, bf16_t* __restrict__ CTX) {
  const int a = blockIdx.y;
  const int q0 = blockIdx.x * 16;
  const int tid = threadIdx.x, wave = tid >> 5, lane = tid & 31;
  const float scale = 0.03125f;  // 1/sqrt(1024) (model dim, per reference)

  __shared__ float part[8][2][256];                      // per-wave exp partials
  __shared__ __align__(16) bf16_t attnb[8][2][16][32];   // per-wave attn tiles

  // Preload Q fragments for this wave's two heads (16 q-rows x 64 d)
  v16bf qf[2][2];
#pragma unroll
  for (int hl = 0; hl < 2; ++hl) {
    int h = wave * 2 + hl;
    const bf16_t* bq = Q + (((size_t)(a * NH + h)) * SEQ + q0) * HD;
    qf[hl][0] = load_frag(bq, HD);
    qf[hl][1] = load_frag(bq + 32, HD);
  }

  v8f o[2][4] = {};  // per head: 16 q x 64 d accumulators

  for (int k0 = 0; k0 < SEQ; k0 += 32) {
    float ex[2][2][8];
#pragma unroll
    for (int hl = 0; hl < 2; ++hl) {
      int h = wave * 2 + hl;
      const bf16_t* bk = Kb + (((size_t)(a * NH + h)) * SEQ + k0) * HD;
      if (k0 + 32 < SEQ) {  // pull next k-tile toward the WGP (global_prefetch_b8)
        __builtin_prefetch(bk + 32 * HD + (lane & 15) * HD, 0, 1);
      }
      v8f sc[2] = {};
#pragma unroll
      for (int g = 0; g < 2; ++g) {  // two 16-token groups of the 32-k tile
        v16bf kf0 = load_frag(bk + g * 16 * HD, HD);
        v16bf kf1 = load_frag(bk + g * 16 * HD + 32, HD);
        sc[g] = WMMA_BF16(qf[hl][0], kf0, sc[g]);
        sc[g] = WMMA_BF16(qf[hl][1], kf1, sc[g]);
      }
#pragma unroll
      for (int g = 0; g < 2; ++g)
#pragma unroll
        for (int r = 0; r < 8; ++r) ex[hl][g][r] = __expf(sc[g][r] * scale);
    }
    // cross-head reduction: each (q,k) element sums exp over all 16 heads
#pragma unroll
    for (int g = 0; g < 2; ++g)
#pragma unroll
      for (int r = 0; r < 8; ++r)
        part[wave][g][r * 32 + lane] = ex[0][g][r] + ex[1][g][r];
    __syncthreads();
    float den[2][8];
#pragma unroll
    for (int g = 0; g < 2; ++g)
#pragma unroll
      for (int r = 0; r < 8; ++r) {
        float s = 0.f;
#pragma unroll
        for (int w = 0; w < 8; ++w) s += part[w][g][r * 32 + lane];
        den[g][r] = s;
      }
    // normalize and repack C-layout -> row-major A-layout tile in LDS
    {
      const int lrow = (lane >> 4) * 8, lcol = lane & 15;
#pragma unroll
      for (int hl = 0; hl < 2; ++hl)
#pragma unroll
        for (int g = 0; g < 2; ++g)
#pragma unroll
          for (int r = 0; r < 8; ++r)
            attnb[wave][hl][r + lrow][g * 16 + lcol] = (bf16_t)(ex[hl][g][r] / den[g][r]);
    }
    __syncthreads();
    // O += attn(16x32) @ V(32x64)
#pragma unroll
    for (int hl = 0; hl < 2; ++hl) {
      int h = wave * 2 + hl;
      v16bf af = load_frag(&attnb[wave][hl][0][0], 32);
      const bf16_t* bv = VT + ((size_t)(a * NH + h)) * HD * SEQ + k0;
      if (k0 + 32 < SEQ) {
        __builtin_prefetch(bv + 32 + (lane & 15) * SEQ, 0, 1);
      }
#pragma unroll
      for (int j = 0; j < 4; ++j) {
        v16bf vf = load_frag(bv + (size_t)(j * 16) * SEQ, SEQ);
        o[hl][j] = WMMA_BF16(af, vf, o[hl][j]);
      }
    }
  }

  // store context in [a][n][h*64+d] so the out-proj GEMM reads it row-major
  const int lrow = (lane >> 4) * 8, lcol = lane & 15;
#pragma unroll
  for (int hl = 0; hl < 2; ++hl) {
    int h = wave * 2 + hl;
#pragma unroll
    for (int j = 0; j < 4; ++j)
#pragma unroll
      for (int r = 0; r < 8; ++r) {
        int qq = q0 + r + lrow;
        int dd = j * 16 + lcol;
        CTX[((size_t)a * SEQ + qq) * 1024 + h * HD + dd] = (bf16_t)o[hl][j][r];
      }
  }
}

// ---------------- host launch ----------------
extern "C" void kernel_launch(void* const* d_in, const int* in_sizes, int n_in,
                              void* d_out, int out_size, void* d_ws, size_t ws_size,
                              hipStream_t stream) {
  const float* x     = (const float*)d_in[0];
  const float* w_qkv = (const float*)d_in[1];
  const float* w_out = (const float*)d_in[2];
  const float* b_out = (const float*)d_in[3];
  float* out = (float*)d_out;
  char* ws = (char*)d_ws;

  const size_t MB = 1ull << 20;
  bf16_t* xb    = (bf16_t*)(ws + 0 * MB);   // 16 MB: x in bf16
  bf16_t* wqkvT = (bf16_t*)(ws + 16 * MB);  //  6 MB: w_qkv^T
  bf16_t* woutT = (bf16_t*)(ws + 22 * MB);  //  2 MB: w_out^T
  bf16_t* Qb    = (bf16_t*)(ws + 24 * MB);  // 16 MB: [a][h][n][64]
  bf16_t* Kbuf  = (bf16_t*)(ws + 40 * MB);  // 16 MB: [a][h][n][64]
  bf16_t* VTb   = (bf16_t*)(ws + 56 * MB);  // 16 MB: [a][h][64][n]
  bf16_t* CTX   = (bf16_t*)(ws + 72 * MB);  // 16 MB: [a][n][1024]

  int nx = 8 * 1024 * 1024;
  cvt_bf16<<<nx / 256, 256, 0, stream>>>(x, xb, nx);
  cvt_bf16_T<<<(1024 * 3072) / 256, 256, 0, stream>>>(w_qkv, wqkvT, 1024, 3072);
  cvt_bf16_T<<<(1024 * 1024) / 256, 256, 0, stream>>>(w_out, woutT, 1024, 1024);

  dim3 g1(3072 / BN, 8192 / BM);
  gemm_bf16<0><<<g1, 128, 0, stream>>>(xb, wqkvT, 8192, 3072, 1024,
                                       Qb, Kbuf, VTb, nullptr, nullptr);

  dim3 g2(SEQ / 16, 8);
  attn_fused<<<g2, 256, 0, stream>>>(Qb, Kbuf, VTb, CTX);

  dim3 g3(1024 / BN, 8192 / BM);
  gemm_bf16<1><<<g3, 128, 0, stream>>>(CTX, woutT, 8192, 1024, 1024,
                                       nullptr, nullptr, nullptr, out, b_out);
}
